// MoELayer_40802189312327
// MI455X (gfx1250) — matrix operand order repaired and checked by reading, not compiled, not run on hardware
//
#include <hip/hip_runtime.h>
#include <math.h>

// ---------------- problem constants ----------------
#define DIM   1024
#define DOUTD 1024
#define NEXP  8
#define HDIM  4096
#define NTOK  4096          // 2*2048
#define TOPK  2
#define AUXC  0.01f

#define KC    256           // K-chunk staged in LDS
#define KC8   (KC / 8)      // 16B groups per row chunk
#define MT    64            // token rows per block
#define NT    256           // output cols per block (2x4 wave grid, 16x128 per wave)

typedef __bf16 bf8   __attribute__((ext_vector_type(8)));
typedef __bf16 v16bf __attribute__((ext_vector_type(16)));
typedef float  v8f   __attribute__((ext_vector_type(8)));

__device__ __forceinline__ unsigned short f2bf(float f) {
    unsigned u = __float_as_uint(f);
    u = u + 0x7FFFu + ((u >> 16) & 1u);     // round-to-nearest-even
    return (unsigned short)(u >> 16);
}
__device__ __forceinline__ unsigned pack2(float a, float b) {
    return (unsigned)f2bf(a) | ((unsigned)f2bf(b) << 16);
}

__device__ __forceinline__ v16bf cat16(bf8 lo, bf8 hi) {
    return __builtin_shufflevector(lo, hi, 0,1,2,3,4,5,6,7,8,9,10,11,12,13,14,15);
}

// ---------------- 1: zero output + counters ----------------
__global__ __launch_bounds__(256) void init_kernel(float* out, int* cnt) {
    size_t i = (size_t)blockIdx.x * 256 + threadIdx.x;
    if (i < (size_t)NTOK * DOUTD) out[i] = 0.0f;
    if (blockIdx.x == 0 && threadIdx.x < NEXP) cnt[threadIdx.x] = 0;
}

// ---------------- 2: fp32 [B][R][C] -> bf16 [B][C][R] (K-major weights) ----------------
__global__ __launch_bounds__(256) void transpose_cast_kernel(const float* __restrict__ in,
                                                             unsigned short* __restrict__ out,
                                                             int R, int C) {
    __shared__ float tile[32][33];
    const size_t bo = (size_t)blockIdx.z * R * C;
    const int c0 = blockIdx.x * 32, r0 = blockIdx.y * 32;
    const int tx = threadIdx.x, ty = threadIdx.y;          // (32,8)
#pragma unroll
    for (int j = 0; j < 32; j += 8)
        tile[ty + j][tx] = in[bo + (size_t)(r0 + ty + j) * C + (c0 + tx)];
    __syncthreads();
#pragma unroll
    for (int j = 0; j < 32; j += 8)
        out[bo + (size_t)(c0 + ty + j) * R + (r0 + tx)] = f2bf(tile[tx][ty + j]);
}

// ---------------- 3: gating (1 wave per token) ----------------
__global__ __launch_bounds__(256) void gating_kernel(const float* __restrict__ x,
                                                     const float* __restrict__ Wg,
                                                     const float* __restrict__ bg,
                                                     int* __restrict__ cnt,
                                                     int* __restrict__ toklist,
                                                     float* __restrict__ wlist,
                                                     float* __restrict__ auxbuf) {
    const int wave = threadIdx.x >> 5;
    const int lane = threadIdx.x & 31;
    const int n = blockIdx.x * 8 + wave;
    if (n >= NTOK) return;

    float acc[8] = {0,0,0,0,0,0,0,0};
    for (int d = lane; d < DIM; d += 32) {
        float xv = x[(size_t)n * DIM + d];
        const float4* w4 = (const float4*)(Wg + (size_t)d * NEXP);
        float4 w0 = w4[0], w1 = w4[1];
        acc[0] += xv * w0.x; acc[1] += xv * w0.y; acc[2] += xv * w0.z; acc[3] += xv * w0.w;
        acc[4] += xv * w1.x; acc[5] += xv * w1.y; acc[6] += xv * w1.z; acc[7] += xv * w1.w;
    }
#pragma unroll
    for (int m = 16; m > 0; m >>= 1) {
#pragma unroll
        for (int e = 0; e < 8; ++e) acc[e] += __shfl_xor(acc[e], m, 32);
    }
    if (lane == 0) {
        float lg[8];
#pragma unroll
        for (int e = 0; e < 8; ++e) lg[e] = acc[e] + bg[e];
        float mx = lg[0];
#pragma unroll
        for (int e = 1; e < 8; ++e) mx = fmaxf(mx, lg[e]);
        float se = 0.f, sl = 0.f;
#pragma unroll
        for (int e = 0; e < 8; ++e) { se += __expf(lg[e] - mx); sl += lg[e]; }
        float lse = __logf(se) + mx;
        auxbuf[n] = sl - 8.f * lse;                 // sum_e log_softmax
        int i1 = 0;
#pragma unroll
        for (int e = 1; e < 8; ++e) if (lg[e] > lg[i1]) i1 = e;
        int i2 = (i1 == 0) ? 1 : 0;
#pragma unroll
        for (int e = 0; e < 8; ++e) if (e != i1 && lg[e] > lg[i2]) i2 = e;
        float g1 = 1.f / (1.f + __expf(lg[i2] - lg[i1]));
        float g2 = 1.f - g1;
        int p1 = atomicAdd(&cnt[i1], 1);
        toklist[i1 * NTOK + p1] = n; wlist[i1 * NTOK + p1] = g1;
        int p2 = atomicAdd(&cnt[i2], 1);
        toklist[i2 * NTOK + p2] = n; wlist[i2 * NTOK + p2] = g2;
    }
}

// ---------------- 4: deterministic aux reduce + count scan ----------------
__global__ __launch_bounds__(256) void finalize_kernel(const float* __restrict__ auxbuf,
                                                       const int* __restrict__ cnt,
                                                       int* __restrict__ base,
                                                       float* __restrict__ aux_out) {
    __shared__ float red[256];
    float s = 0.f;
    for (int i = threadIdx.x; i < NTOK; i += 256) s += auxbuf[i];
    red[threadIdx.x] = s;
    __syncthreads();
    for (int off = 128; off > 0; off >>= 1) {
        if (threadIdx.x < off) red[threadIdx.x] += red[threadIdx.x + off];
        __syncthreads();
    }
    if (threadIdx.x == 0) {
        const float invE = 1.f / NEXP;
        aux_out[0] = AUXC * (invE * logf(invE) - red[0] / ((float)NEXP * NEXP * NTOK));
        int b = 0;
        for (int e = 0; e < NEXP; ++e) { base[e] = b; b += cnt[e]; }
    }
}

// ---------------- WMMA fragment helpers ----------------
__device__ __forceinline__ v16bf load_a_frag(const unsigned short* arow, int ka) {
    bf8 lo = *(const bf8*)(arow + ka);
    bf8 hi = *(const bf8*)(arow + ka + 16);
    return cat16(lo, hi);
}
__device__ __forceinline__ v16bf load_b_frag(const unsigned short* bcol, int kb) {
    bf8 lo = *(const bf8*)(bcol + kb);
    bf8 hi = *(const bf8*)(bcol + kb + 16);
    return cat16(lo, hi);
}

// ---------------- 5: expert GEMM1: h = gelu(x @ W1 + b1), bf16 out ----------------
__global__ __launch_bounds__(256) void moe_gemm1_kernel(const float* __restrict__ x,
                                                        const unsigned short* __restrict__ W1bf,
                                                        const float* __restrict__ b1,
                                                        const int* __restrict__ cnt,
                                                        const int* __restrict__ base,
                                                        const int* __restrict__ toklist,
                                                        unsigned short* __restrict__ hbuf) {
    const int e = blockIdx.z;
    const int cnt_e = cnt[e];
    const int m0 = blockIdx.x * MT;
    if (m0 >= cnt_e) return;
    const int base_e = base[e];

    const int tid  = threadIdx.x;
    const int wave = tid >> 5, lane = tid & 31;
    const int mband = wave & 3;          // 4 x 16 rows
    const int nband = wave >> 2;         // 2 x 128 cols
    const int n0 = blockIdx.y * NT + nband * 128;

    __shared__ unsigned short Xs[MT][KC + 8];
    __shared__ int tok_s[MT];

    if (tid < MT) {
        int pos = m0 + tid;
        tok_s[tid] = (pos < cnt_e) ? toklist[e * NTOK + pos] : 0;
    }
    __syncthreads();

    v8f acc[8];
#pragma unroll
    for (int t = 0; t < 8; ++t) acc[t] = (v8f)(0.f);

    const int colbase = n0 + (lane & 15);
    const unsigned short* bptr[8];
#pragma unroll
    for (int t = 0; t < 8; ++t)
        bptr[t] = W1bf + ((size_t)e * HDIM + colbase + t * 16) * DIM;

    const unsigned short* arow = &Xs[mband * 16 + (lane & 15)][0];
    const int kb_a = (lane & 16) ? 8 : 0;
    const int kb_b = (lane & 16) ? 8 : 0;

    for (int kc = 0; kc < DIM; kc += KC) {
        // vectorized gather+convert: 8 bf16 (one ds_store_b128) per iteration
#pragma unroll
        for (int idx = tid; idx < MT * KC8; idx += 256) {
            int r  = idx >> 5;                    // idx / KC8
            int c8 = (idx & (KC8 - 1)) * 8;
            const float4* xp = (const float4*)(x + (size_t)tok_s[r] * DIM + kc + c8);
            float4 a0 = xp[0], a1 = xp[1];
            uint4 pk;
            pk.x = pack2(a0.x, a0.y);
            pk.y = pack2(a0.z, a0.w);
            pk.z = pack2(a1.x, a1.y);
            pk.w = pack2(a1.z, a1.w);
            *(uint4*)&Xs[r][c8] = pk;
        }
        __syncthreads();
#pragma unroll
        for (int kk = 0; kk < KC / 32; ++kk) {
            v16bf a = load_a_frag(arow, kk * 32 + kb_a);
            const int kg = kc + kk * 32 + kb_b;
#pragma unroll
            for (int t = 0; t < 8; ++t) {
                v16bf b = load_b_frag(bptr[t], kg);
                acc[t] = __builtin_amdgcn_wmma_f32_16x16x32_bf16(
                    false, a, false, b, (short)0, acc[t], false, false);
            }
        }
        __syncthreads();
    }

    const int col = lane & 15;
    const int rbase = mband * 16 + ((lane & 16) ? 8 : 0);
#pragma unroll
    for (int t = 0; t < 8; ++t) {
        const int hcol = n0 + t * 16 + col;
        const float bias = b1[e * HDIM + hcol];
#pragma unroll
        for (int vr = 0; vr < 8; ++vr) {
            int pos = m0 + rbase + vr;
            if (pos < cnt_e) {
                float v = acc[t][vr] + bias;
                v = 0.5f * v * (1.f + erff(v * 0.70710678118654752f));   // exact GELU
                hbuf[(size_t)(base_e + pos) * HDIM + hcol] = f2bf(v);
            }
        }
    }
}

// ---------------- 6: expert GEMM2: out += w * (h @ W2 + b2) ----------------
__global__ __launch_bounds__(256) void moe_gemm2_kernel(const unsigned short* __restrict__ hbuf,
                                                        const unsigned short* __restrict__ W2bf,
                                                        const float* __restrict__ b2,
                                                        const int* __restrict__ cnt,
                                                        const int* __restrict__ base,
                                                        const int* __restrict__ toklist,
                                                        const float* __restrict__ wlist,
                                                        float* __restrict__ out) {
    const int e = blockIdx.z;
    const int cnt_e = cnt[e];
    const int m0 = blockIdx.x * MT;
    if (m0 >= cnt_e) return;
    const int base_e = base[e];

    const int tid  = threadIdx.x;
    const int wave = tid >> 5, lane = tid & 31;
    const int mband = wave & 3;
    const int nband = wave >> 2;
    const int n0 = blockIdx.y * NT + nband * 128;

    __shared__ unsigned short Xs[MT][KC + 8];
    __shared__ int   tok_s[MT];
    __shared__ float w_s[MT];

    if (tid < MT) {
        int pos = m0 + tid;
        bool v = pos < cnt_e;
        tok_s[tid] = v ? toklist[e * NTOK + pos] : 0;
        w_s[tid]   = v ? wlist[e * NTOK + pos] : 0.f;
    }
    __syncthreads();

    v8f acc[8];
#pragma unroll
    for (int t = 0; t < 8; ++t) acc[t] = (v8f)(0.f);

    const int colbase = n0 + (lane & 15);
    const unsigned short* bptr[8];
#pragma unroll
    for (int t = 0; t < 8; ++t)
        bptr[t] = W2bf + ((size_t)e * DOUTD + colbase + t * 16) * HDIM;

    const unsigned short* arow = &Xs[mband * 16 + (lane & 15)][0];
    const int kb_a = (lane & 16) ? 8 : 0;
    const int kb_b = (lane & 16) ? 8 : 0;

    for (int kc = 0; kc < HDIM; kc += KC) {
        // vectorized copy: 8 bf16 (16B global load + 16B ds store) per iteration
#pragma unroll
        for (int idx = tid; idx < MT * KC8; idx += 256) {
            int r  = idx >> 5;
            int c8 = (idx & (KC8 - 1)) * 8;
            int pos = m0 + r;
            uint4 pk = make_uint4(0u, 0u, 0u, 0u);
            if (pos < cnt_e)
                pk = *(const uint4*)(hbuf + (size_t)(base_e + pos) * HDIM + kc + c8);
            *(uint4*)&Xs[r][c8] = pk;
        }
        __syncthreads();
#pragma unroll
        for (int kk = 0; kk < KC / 32; ++kk) {
            v16bf a = load_a_frag(arow, kk * 32 + kb_a);
            const int kg = kc + kk * 32 + kb_b;
#pragma unroll
            for (int t = 0; t < 8; ++t) {
                v16bf b = load_b_frag(bptr[t], kg);
                acc[t] = __builtin_amdgcn_wmma_f32_16x16x32_bf16(
                    false, a, false, b, (short)0, acc[t], false, false);
            }
        }
        __syncthreads();
    }

    const int col = lane & 15;
    const int rbase = mband * 16 + ((lane & 16) ? 8 : 0);
#pragma unroll
    for (int t = 0; t < 8; ++t) {
        const int dcol = n0 + t * 16 + col;
        const float bias = b2[e * DOUTD + dcol];
#pragma unroll
        for (int vr = 0; vr < 8; ++vr) {
            int pos = m0 + rbase + vr;
            if (pos < cnt_e) {
                int r = pos - m0;
                float v = acc[t][vr] + bias;
                atomicAdd(&out[(size_t)tok_s[r] * DOUTD + dcol], w_s[r] * v);
            }
        }
    }
}

// ---------------- launcher ----------------
extern "C" void kernel_launch(void* const* d_in, const int* in_sizes, int n_in,
                              void* d_out, int out_size, void* d_ws, size_t ws_size,
                              hipStream_t stream) {
    const float* x  = (const float*)d_in[0];
    const float* Wg = (const float*)d_in[1];
    const float* bg = (const float*)d_in[2];
    const float* W1 = (const float*)d_in[3];
    const float* b1 = (const float*)d_in[4];
    const float* W2 = (const float*)d_in[5];
    const float* b2 = (const float*)d_in[6];
    float* out = (float*)d_out;

    // workspace layout
    char* ws = (char*)d_ws;
    const size_t W1BF_B = (size_t)NEXP * HDIM * DIM * 2;       // 64MB
    const size_t W2BF_B = (size_t)NEXP * DOUTD * HDIM * 2;     // 64MB
    const size_t HBUF_B = (size_t)NTOK * TOPK * HDIM * 2;      // 64MB
    unsigned short* W1bf = (unsigned short*)ws;
    unsigned short* W2bf = (unsigned short*)(ws + W1BF_B);
    unsigned short* hbuf = (unsigned short*)(ws + W1BF_B + W2BF_B);
    char* p = ws + W1BF_B + W2BF_B + HBUF_B;
    int*   toklist = (int*)p;                 p += (size_t)NEXP * NTOK * 4;
    float* wlist   = (float*)p;               p += (size_t)NEXP * NTOK * 4;
    float* auxbuf  = (float*)p;               p += (size_t)NTOK * 4;
    int*   cnt     = (int*)p;
    int*   base    = cnt + NEXP;

    // 1. zero out + counters
    init_kernel<<<(NTOK * DOUTD) / 256, 256, 0, stream>>>(out, cnt);

    // 2. weights -> bf16, K-major
    transpose_cast_kernel<<<dim3(HDIM / 32, DIM / 32, NEXP), dim3(32, 8), 0, stream>>>(
        W1, W1bf, DIM, HDIM);
    transpose_cast_kernel<<<dim3(DOUTD / 32, HDIM / 32, NEXP), dim3(32, 8), 0, stream>>>(
        W2, W2bf, HDIM, DOUTD);

    // 3. gating / routing
    gating_kernel<<<NTOK / 8, 256, 0, stream>>>(x, Wg, bg, cnt, toklist, wlist, auxbuf);

    // 4. aux loss + prefix scan
    finalize_kernel<<<1, 256, 0, stream>>>(auxbuf, cnt, base, out + (size_t)NTOK * DOUTD);

    // 5. expert FFN layer 1
    moe_gemm1_kernel<<<dim3(NTOK / MT, HDIM / NT, NEXP), 256, 0, stream>>>(
        x, W1bf, b1, cnt, base, toklist, hbuf);

    // 6. expert FFN layer 2 + weighted combine
    moe_gemm2_kernel<<<dim3(NTOK / MT, DOUTD / NT, NEXP), 256, 0, stream>>>(
        hbuf, W2bf, b2, cnt, base, toklist, wlist, out);
}
